// MaskedCrossAttention_1726576857555
// MI455X (gfx1250) — matrix operand order repaired and checked by reading, not compiled, not run on hardware
//
#include <hip/hip_runtime.h>
#include <hip/hip_bf16.h>

typedef __bf16 bf16;
typedef __attribute__((ext_vector_type(16))) __bf16 bf16x16;
typedef __attribute__((ext_vector_type(8)))  __bf16 bf16x8;
typedef __attribute__((ext_vector_type(2)))  __bf16 bf16x2;
typedef __attribute__((ext_vector_type(8)))  float  f32x8;

#define BB      4
#define T_TXT   2048
#define DIM     2048
#define DIM_VIS 1024
#define INNER   1024
#define HEADS   16
#define DHEAD   64
#define T_IMG   32
#define N_LAT   64
#define NMEDIA  (T_IMG * N_LAT)   // 2048 KV tokens per batch
#define ROWS    (BB * T_TXT)      // 8192

// gfx1250 async global->LDS copy (ASYNCcnt-tracked). The builtin's parameter
// types (from the compiler diagnostic) are:
//   arg0: int __attribute__((vector_size(16))) * in addrspace(1)  (global src)
//   arg1: LDS destination pointer                                  (local dst)
//   arg2/arg3: immediate offset / cache-policy
#if defined(__has_builtin)
#if __has_builtin(__builtin_amdgcn_global_load_async_to_lds_b128) && \
    __has_builtin(__builtin_amdgcn_s_wait_asynccnt)
#define USE_ASYNC_LDS 1
#endif
#endif

#if defined(USE_ASYNC_LDS)
typedef int i32x4 __attribute__((vector_size(16)));
typedef __attribute__((address_space(1))) i32x4 as1_i32x4;
typedef __attribute__((address_space(3))) i32x4 as3_i32x4;
#define ASYNC_COPY_B128(gsrc, ldst) \
    __builtin_amdgcn_global_load_async_to_lds_b128((as1_i32x4*)(gsrc), (as3_i32x4*)(ldst), 0, 0)
#define WAIT_ASYNC() __builtin_amdgcn_s_wait_asynccnt(0)
#endif

// ---------------------------------------------------------------------------
// 1) per-row inclusive cumsum of media_locations (bool/uint8) -> int32
// ---------------------------------------------------------------------------
__global__ __launch_bounds__(128) void cumsum_kernel(const unsigned char* __restrict__ loc,
                                                     int* __restrict__ tt) {
    const int row  = threadIdx.x >> 5;   // 0..3
    const int lane = threadIdx.x & 31;
    const unsigned char* lr = loc + (size_t)row * T_TXT;
    int* tr = tt + (size_t)row * T_TXT;
    int offset = 0;
    for (int c = 0; c < T_TXT; c += 32) {
        int v = lr[c + lane] ? 1 : 0;
        int scan = v;
        #pragma unroll
        for (int o = 1; o < 32; o <<= 1) {
            int nv = __shfl_up(scan, o, 32);
            if (lane >= o) scan += nv;
        }
        tr[c + lane] = offset + scan;
        offset += __shfl(scan, 31, 32);
    }
}

// ---------------------------------------------------------------------------
// 2) fused LayerNorm (eps=1e-5) with bf16 output. One 256-thread block per row.
// ---------------------------------------------------------------------------
__global__ __launch_bounds__(256) void ln_kernel(const float* __restrict__ x,
                                                 const float* __restrict__ g,
                                                 const float* __restrict__ beta,
                                                 bf16* __restrict__ xn) {
    const int row = blockIdx.x;
    const float* xr = x + (size_t)row * DIM;
    float s = 0.f, s2 = 0.f;
    for (int c = threadIdx.x; c < DIM; c += 256) {
        float v = xr[c];
        s += v; s2 += v * v;
    }
    #pragma unroll
    for (int o = 16; o > 0; o >>= 1) {
        s  += __shfl_xor(s,  o, 32);
        s2 += __shfl_xor(s2, o, 32);
    }
    __shared__ float red[2][8];
    const int wave = threadIdx.x >> 5, lane = threadIdx.x & 31;
    if (lane == 0) { red[0][wave] = s; red[1][wave] = s2; }
    __syncthreads();
    float ts = 0.f, ts2 = 0.f;
    #pragma unroll
    for (int w = 0; w < 8; ++w) { ts += red[0][w]; ts2 += red[1][w]; }
    const float mu  = ts * (1.f / DIM);
    const float var = ts2 * (1.f / DIM) - mu * mu;
    const float rs  = rsqrtf(var + 1e-5f);
    bf16* xo = xn + (size_t)row * DIM;
    for (int c = threadIdx.x; c < DIM; c += 256)
        xo[c] = (bf16)((xr[c] - mu) * rs * g[c] + beta[c]);
}

// ---------------------------------------------------------------------------
// 3) elementwise f32 -> bf16 (sizes are multiples of 256)
// ---------------------------------------------------------------------------
__global__ __launch_bounds__(256) void cvt_bf16_kernel(const float* __restrict__ in,
                                                       bf16* __restrict__ out) {
    const size_t i = (size_t)blockIdx.x * 256 + threadIdx.x;
    out[i] = (bf16)in[i];
}

// ---------------------------------------------------------------------------
// 4) tiled bf16 WMMA GEMM: C[MxN] = A[MxK] * B[KxN], A/B bf16 row-major.
//    Block tile 128x128, K-step 32, 8 waves (4 along M x 2 along N), each wave
//    computing 2x4 16x16 tiles with v_wmma_f32_16x16x32_bf16.
//    Software pipelined with double-buffered LDS:
//      - A tile goes global->LDS via GLOBAL_LOAD_ASYNC_TO_LDS_B128 (ASYNCcnt)
//        when available, else register-staged b128 copies.
//      - B tile is register-staged and transposed into LDS [n][k] so B
//        fragments load as two ds_load_b128 (documented CDNA5 B striping).
//    Next tile's loads are issued BEFORE the current tile's WMMAs so global
//    latency hides behind matrix math.
// ---------------------------------------------------------------------------
template <typename OutT>
__global__ __launch_bounds__(256) void gemm_bf16_wmma(const bf16* __restrict__ A,
                                                      const bf16* __restrict__ B,
                                                      OutT* __restrict__ C,
                                                      int M, int N, int K) {
    constexpr int BM = 128, BN = 128, BK = 32;
    constexpr int LDA = BK + 16;   // 48 elems = 96 B row stride, 16B aligned frags
    constexpr int LDB = BK + 16;
    __shared__ bf16 Alds[2][BM * LDA];
    __shared__ bf16 Blds[2][BN * LDB];

    const int tid   = threadIdx.x;
    const int lane  = tid & 31;
    const int wave  = tid >> 5;     // 0..7
    const int waveM = wave & 3;     // 4 waves along M
    const int waveN = wave >> 2;    // 2 waves along N
    const int blockM = blockIdx.y * BM;
    const int blockN = blockIdx.x * BN;

    f32x8 acc[2][4];
    #pragma unroll
    for (int mt = 0; mt < 2; ++mt)
        #pragma unroll
        for (int nt = 0; nt < 4; ++nt)
            #pragma unroll
            for (int r = 0; r < 8; ++r)
                acc[mt][nt][r] = 0.f;

    const int fm    = lane & 15;          // m (A frag) / n (B frag) within tile
    const int koffA = (lane >> 4) * 8;    // A K-half select
    const int kbB   = (lane >> 4) * 8;    // B K-half select

    // global->LDS staging coordinates
    const int ar = tid >> 2;              // A: row 0..63 (+64 second half)
    const int ac = (tid & 3) * 8;         // A: col 0,8,16,24
    const int bk = tid >> 4;              // B: k 0..15 (+16 second half)
    const int bn = (tid & 15) * 8;        // B: n chunk of 8

    const bf16* Asrc0 = A + (size_t)(blockM + ar) * K + ac;
    const bf16* Asrc1 = A + (size_t)(blockM + ar + 64) * K + ac;
    const bf16* Bsrc0 = B + (size_t)bk * N + blockN + bn;
    const bf16* Bsrc1 = B + (size_t)(bk + 16) * N + blockN + bn;

    // ---- prologue: stage tile 0 into buffer 0 ----
    {
#if defined(USE_ASYNC_LDS)
        ASYNC_COPY_B128(Asrc0, &Alds[0][ar * LDA + ac]);
        ASYNC_COPY_B128(Asrc1, &Alds[0][(ar + 64) * LDA + ac]);
#else
        bf16x8 a0 = *(const bf16x8*)Asrc0;
        bf16x8 a1 = *(const bf16x8*)Asrc1;
#endif
        bf16x8 b0 = *(const bf16x8*)Bsrc0;
        bf16x8 b1 = *(const bf16x8*)Bsrc1;
#if !defined(USE_ASYNC_LDS)
        *(bf16x8*)&Alds[0][ar * LDA + ac]        = a0;
        *(bf16x8*)&Alds[0][(ar + 64) * LDA + ac] = a1;
#endif
        #pragma unroll
        for (int e = 0; e < 8; ++e) {
            Blds[0][(bn + e) * LDB + bk]      = b0[e];
            Blds[0][(bn + e) * LDB + bk + 16] = b1[e];
        }
#if defined(USE_ASYNC_LDS)
        WAIT_ASYNC();
#endif
    }
    __syncthreads();

    int buf = 0;
    for (int kk = 0; kk < K; kk += BK) {
        const int  nbuf     = buf ^ 1;
        const bool has_next = (kk + BK) < K;

        // ---- issue next tile's global loads (not waited) ----
        bf16x8 b0n, b1n;
#if !defined(USE_ASYNC_LDS)
        bf16x8 a0n, a1n;
#endif
        if (has_next) {
            const size_t ko = (size_t)(kk + BK);
#if defined(USE_ASYNC_LDS)
            ASYNC_COPY_B128(Asrc0 + ko, &Alds[nbuf][ar * LDA + ac]);
            ASYNC_COPY_B128(Asrc1 + ko, &Alds[nbuf][(ar + 64) * LDA + ac]);
#else
            a0n = *(const bf16x8*)(Asrc0 + ko);
            a1n = *(const bf16x8*)(Asrc1 + ko);
#endif
            b0n = *(const bf16x8*)(Bsrc0 + ko * N);
            b1n = *(const bf16x8*)(Bsrc1 + ko * N);
        }

        // ---- compute on current buffer ----
        const bf16* Ab = Alds[buf];
        const bf16* Bb = Blds[buf];

        bf16x16 afrag[2];
        #pragma unroll
        for (int mt = 0; mt < 2; ++mt) {
            const int m = waveM * 32 + mt * 16 + fm;
            bf16x8 lo = *(const bf16x8*)&Ab[m * LDA + koffA];
            bf16x8 hi = *(const bf16x8*)&Ab[m * LDA + 16 + koffA];
            #pragma unroll
            for (int e = 0; e < 8; ++e) { afrag[mt][e] = lo[e]; afrag[mt][8 + e] = hi[e]; }
        }
        bf16x16 bfrag[4];
        #pragma unroll
        for (int nt = 0; nt < 4; ++nt) {
            const int n = waveN * 64 + nt * 16 + fm;
            bf16x8 lo = *(const bf16x8*)&Bb[n * LDB + kbB];
            bf16x8 hi = *(const bf16x8*)&Bb[n * LDB + 16 + kbB];
            #pragma unroll
            for (int e = 0; e < 8; ++e) { bfrag[nt][e] = lo[e]; bfrag[nt][8 + e] = hi[e]; }
        }
        #pragma unroll
        for (int mt = 0; mt < 2; ++mt)
            #pragma unroll
            for (int nt = 0; nt < 4; ++nt)
                acc[mt][nt] = __builtin_amdgcn_wmma_f32_16x16x32_bf16(
                    false, afrag[mt], false, bfrag[nt],
                    (short)0, acc[mt][nt], false, false);

        // ---- commit next tile into its LDS buffer ----
        if (has_next) {
#if !defined(USE_ASYNC_LDS)
            *(bf16x8*)&Alds[nbuf][ar * LDA + ac]        = a0n;
            *(bf16x8*)&Alds[nbuf][(ar + 64) * LDA + ac] = a1n;
#endif
            #pragma unroll
            for (int e = 0; e < 8; ++e) {
                Blds[nbuf][(bn + e) * LDB + bk]      = b0n[e];
                Blds[nbuf][(bn + e) * LDB + bk + 16] = b1n[e];
            }
#if defined(USE_ASYNC_LDS)
            WAIT_ASYNC();
#endif
        }
        __syncthreads();
        buf = nbuf;
    }

    const int cn = lane & 15;
    const int cm = (lane >> 4) * 8;
    #pragma unroll
    for (int mt = 0; mt < 2; ++mt)
        #pragma unroll
        for (int nt = 0; nt < 4; ++nt) {
            const int gm = blockM + waveM * 32 + mt * 16 + cm;
            const int gn = blockN + waveN * 64 + nt * 16 + cn;
            #pragma unroll
            for (int r = 0; r < 8; ++r)
                C[(size_t)(gm + r) * N + gn] = (OutT)acc[mt][nt][r];
        }
}

// ---------------------------------------------------------------------------
// 5) per-(b,head) v-mean for the text_time > 32 branch (uniform attention)
// ---------------------------------------------------------------------------
__global__ __launch_bounds__(256) void vmean_kernel(const bf16* __restrict__ kv,
                                                    float* __restrict__ vmean) {
    const int b = blockIdx.x >> 2;
    const int d = (blockIdx.x & 3) * 256 + threadIdx.x;   // 0..1023
    const bf16* base = kv + (size_t)b * NMEDIA * (2 * INNER) + INNER + d;
    float s = 0.f;
    for (int j = 0; j < NMEDIA; ++j) s += (float)base[(size_t)j * (2 * INNER)];
    vmean[b * INNER + d] = s * (1.f / NMEDIA);
}

// ---------------------------------------------------------------------------
// 6) block-sparse masked attention: text token with text_time t in [1,32]
//    attends exactly to media block (t-1): 64 keys, 64-wide softmax.
//    t==0 -> zero row; t>32 -> uniform attention == mean of v.
//    One wave per head; shuffle-only reductions, no barriers.
// ---------------------------------------------------------------------------
__global__ __launch_bounds__(512) void attn_kernel(const bf16* __restrict__ q,      // [8192][1024]
                                                   const bf16* __restrict__ kv,     // [8192][2048]
                                                   const int* __restrict__ tt,      // [4][2048]
                                                   const float* __restrict__ vmean, // [4][1024]
                                                   bf16* __restrict__ out) {        // [8192][1024]
    const int token = blockIdx.x;            // b*T_TXT + i
    const int b     = token >> 11;
    const int h     = threadIdx.x >> 5;
    const int lane  = threadIdx.x & 31;
    const int t     = tt[token];

    bf16* oh = out + (size_t)token * INNER + h * DHEAD;

    if (t == 0) {
        bf16x2 z; z[0] = (bf16)0.f; z[1] = (bf16)0.f;
        *(bf16x2*)&oh[lane * 2] = z;
        return;
    }
    if (t > T_IMG) {
        const float* vm = vmean + b * INNER + h * DHEAD;
        oh[lane * 2]     = (bf16)vm[lane * 2];
        oh[lane * 2 + 1] = (bf16)vm[lane * 2 + 1];
        return;
    }

    const bf16* qh = q + (size_t)token * INNER + h * DHEAD;
    const size_t jbase = (size_t)b * NMEDIA + (size_t)(t - 1) * 64;
    const bf16* kb = kv + jbase * (2 * INNER) + h * DHEAD;
    const bf16* vb = kb + INNER;

    float s0 = 0.f, s1 = 0.f;
    #pragma unroll
    for (int c = 0; c < 8; ++c) {
        bf16x8 qv = *(const bf16x8*)&qh[c * 8];
        bf16x8 k0 = *(const bf16x8*)&kb[(size_t)lane        * (2 * INNER) + c * 8];
        bf16x8 k1 = *(const bf16x8*)&kb[(size_t)(lane + 32) * (2 * INNER) + c * 8];
        #pragma unroll
        for (int e = 0; e < 8; ++e) {
            const float qe = (float)qv[e];
            s0 += qe * (float)k0[e];
            s1 += qe * (float)k1[e];
        }
    }
    const float scale = 0.125f;   // DIM_HEAD^-0.5
    s0 *= scale; s1 *= scale;

    float m = fmaxf(s0, s1);
    #pragma unroll
    for (int o = 16; o > 0; o >>= 1) m = fmaxf(m, __shfl_xor(m, o, 32));
    float p0 = __expf(s0 - m), p1 = __expf(s1 - m);
    float sum = p0 + p1;
    #pragma unroll
    for (int o = 16; o > 0; o >>= 1) sum += __shfl_xor(sum, o, 32);
    const float inv = 1.f / sum;
    p0 *= inv; p1 *= inv;

    float o0 = 0.f, o1 = 0.f;
    #pragma unroll 4
    for (int src = 0; src < 32; ++src) {
        const float pj0 = __shfl(p0, src, 32);
        const float pj1 = __shfl(p1, src, 32);
        bf16x2 va = *(const bf16x2*)&vb[(size_t)src        * (2 * INNER) + lane * 2];
        bf16x2 vc = *(const bf16x2*)&vb[(size_t)(src + 32) * (2 * INNER) + lane * 2];
        o0 += pj0 * (float)va[0] + pj1 * (float)vc[0];
        o1 += pj0 * (float)va[1] + pj1 * (float)vc[1];
    }
    oh[lane * 2]     = (bf16)o0;
    oh[lane * 2 + 1] = (bf16)o1;
}

// ---------------------------------------------------------------------------
// launch
// ---------------------------------------------------------------------------
extern "C" void kernel_launch(void* const* d_in, const int* in_sizes, int n_in,
                              void* d_out, int out_size, void* d_ws, size_t ws_size,
                              hipStream_t stream) {
    const float* x     = (const float*)d_in[0];
    const float* media = (const float*)d_in[1];  // (B, 32, 64, 1024) == (B, 2048, 1024)
    const unsigned char* mloc = (const unsigned char*)d_in[2];
    const float* ln_g  = (const float*)d_in[3];
    const float* ln_b  = (const float*)d_in[4];
    const float* Wq    = (const float*)d_in[5];
    const float* Wkv   = (const float*)d_in[6];
    const float* Wout  = (const float*)d_in[7];
    float* out = (float*)d_out;

    // workspace carve-up (256B aligned)
    char* ws = (char*)d_ws;
    auto alloc = [&](size_t bytes) {
        void* p = (void*)ws;
        ws += (bytes + 255) & ~(size_t)255;
        return p;
    };
    int*   tt       = (int*)  alloc((size_t)ROWS * 4);
    bf16*  xn_bf    = (bf16*) alloc((size_t)ROWS * DIM * 2);
    bf16*  media_bf = (bf16*) alloc((size_t)ROWS * DIM_VIS * 2);
    bf16*  Wq_bf    = (bf16*) alloc((size_t)DIM * INNER * 2);
    bf16*  Wkv_bf   = (bf16*) alloc((size_t)DIM_VIS * 2 * INNER * 2);
    bf16*  Wout_bf  = (bf16*) alloc((size_t)INNER * DIM * 2);
    bf16*  q_bf     = (bf16*) alloc((size_t)ROWS * INNER * 2);
    bf16*  kv_bf    = (bf16*) alloc((size_t)ROWS * 2 * INNER * 2);
    bf16*  attn_bf  = (bf16*) alloc((size_t)ROWS * INNER * 2);
    float* vmean    = (float*)alloc((size_t)BB * INNER * 4);

    // 1) text_time
    cumsum_kernel<<<1, 128, 0, stream>>>(mloc, tt);
    // 2) LayerNorm -> bf16
    ln_kernel<<<ROWS, 256, 0, stream>>>(x, ln_g, ln_b, xn_bf);
    // 3) converts
    cvt_bf16_kernel<<<(ROWS * DIM_VIS) / 256, 256, 0, stream>>>(media, media_bf);
    cvt_bf16_kernel<<<(DIM * INNER) / 256, 256, 0, stream>>>(Wq, Wq_bf);
    cvt_bf16_kernel<<<(DIM_VIS * 2 * INNER) / 256, 256, 0, stream>>>(Wkv, Wkv_bf);
    cvt_bf16_kernel<<<(INNER * DIM) / 256, 256, 0, stream>>>(Wout, Wout_bf);
    // 4) q = xn @ Wq   (8192x2048 * 2048x1024 -> bf16)
    gemm_bf16_wmma<bf16><<<dim3(INNER / 128, ROWS / 128), 256, 0, stream>>>(
        xn_bf, Wq_bf, q_bf, ROWS, INNER, DIM);
    // 5) kv = media @ Wkv (8192x1024 * 1024x2048 -> bf16)
    gemm_bf16_wmma<bf16><<<dim3((2 * INNER) / 128, ROWS / 128), 256, 0, stream>>>(
        media_bf, Wkv_bf, kv_bf, ROWS, 2 * INNER, DIM_VIS);
    // 6) v-mean (for the all-masked uniform-softmax edge case)
    vmean_kernel<<<BB * 4, 256, 0, stream>>>(kv_bf, vmean);
    // 7) block-sparse masked attention
    attn_kernel<<<ROWS, 512, 0, stream>>>(q_bf, kv_bf, tt, vmean, attn_bf);
    // 8) out = attn_out @ Wout (8192x1024 * 1024x2048 -> f32 d_out)
    gemm_bf16_wmma<float><<<dim3(DIM / 128, ROWS / 128), 256, 0, stream>>>(
        attn_bf, Wout_bf, out, ROWS, DIM, INNER);
}
